// CGCN_39161511805531
// MI455X (gfx1250) — compile-verified, matrix operand order checked by reading, not run on hardware
//
#include <hip/hip_runtime.h>
#include <hip/hip_bf16.h>

typedef __bf16 bf16_t;
typedef __bf16 v16bf __attribute__((ext_vector_type(16)));
typedef float  v8f   __attribute__((ext_vector_type(8)));

#define WMMA_BF16(a, b, c) \
  __builtin_amdgcn_wmma_f32_16x16x32_bf16(false, (a), false, (b), (short)0, (c), false, false)

// ---------------------------------------------------------------------------
// 1) edge weight: ew[e] = lin1_b + sum_k lin1_w[k] * (1 + exp(-c*(k+1)))/2
// ---------------------------------------------------------------------------
__global__ void k_edge_weight(const float* __restrict__ curv,
                              const float* __restrict__ lw,
                              const float* __restrict__ lb,
                              float* __restrict__ ew, int E) {
  int e = blockIdx.x * blockDim.x + threadIdx.x;
  if (e >= E) return;
  float c = curv[e];
  float acc = lb[0];
#pragma unroll
  for (int k = 0; k < 10; ++k)
    acc += lw[k] * (1.0f + __expf(-c * (float)(k + 1))) * 0.5f;
  ew[e] = acc;
}

__global__ void k_fill(float* __restrict__ p, float v, long long n) {
  long long i = (long long)blockIdx.x * blockDim.x + threadIdx.x;
  if (i < n) p[i] = v;
}

// deg[col[e]] += ew[e]  (deg pre-initialized to 1.0 = self-loop weight)
__global__ void k_deg(const int* __restrict__ col, const float* __restrict__ ew,
                      float* __restrict__ deg, int E) {
  int e = blockIdx.x * blockDim.x + threadIdx.x;
  if (e < E) atomicAdd(&deg[col[e]], ew[e]);
}

__global__ void k_dinv(const float* __restrict__ deg, float* __restrict__ dinv, int N) {
  int i = blockIdx.x * blockDim.x + threadIdx.x;
  if (i >= N) return;
  float d = deg[i];
  dinv[i] = (d > 0.0f) ? rsqrtf(d) : 0.0f;
}

// norm[e] = dinv[row]*ew[e]*dinv[col]  (in-place over ew)
__global__ void k_norm(const int* __restrict__ eidx, const float* __restrict__ dinv,
                       float* __restrict__ ew, int E) {
  int e = blockIdx.x * blockDim.x + threadIdx.x;
  if (e >= E) return;
  int r = eidx[e], c = eidx[E + e];
  ew[e] = dinv[r] * ew[e] * dinv[c];
}

// ---------------------------------------------------------------------------
// 2) pack f32 weight [K, Nn] into bf16 WMMA B-fragments.
//    frag = ntile*(K/32) + kstep ; per frag: 32 lanes x 16 bf16 contiguous.
//    lane L: n = ntile*16 + (L&15); element e: k = kstep*32 + (e&7) + 16*(e>>3) + 8*(L>>4)
// ---------------------------------------------------------------------------
__global__ void k_pack_w(const float* __restrict__ W, bf16_t* __restrict__ dst,
                         int K, int Nn) {
  int t = blockIdx.x * blockDim.x + threadIdx.x;
  int ksteps = K >> 5;
  int frags = (Nn >> 4) * ksteps;
  if (t >= frags * 32) return;
  int lane = t & 31;
  int frag = t >> 5;
  int ntile = frag / ksteps;
  int kstep = frag % ksteps;
  int n = ntile * 16 + (lane & 15);
  int hi = lane >> 4;
  bf16_t* o = dst + ((size_t)frag * 32 + lane) * 16;
#pragma unroll
  for (int e = 0; e < 16; ++e) {
    int k = kstep * 32 + (e & 7) + ((e >> 3) << 4) + 8 * hi;
    o[e] = (bf16_t)W[(size_t)k * Nn + n];
  }
}

// ---------------------------------------------------------------------------
// 3) xw = x @ W_gcn : M=N_nodes, K=256, Nn=128. Block: 256 thr (8 waves),
//    tile 64(M) x 128(N); wave (wm,wn) in 2x4 grid computes 32x32 (2x2 WMMA).
// ---------------------------------------------------------------------------
#define XS 264  // LDS row stride in bf16 (132 dwords == 4 mod 64 banks)
__global__ void k_gemm_xw(const float* __restrict__ x, const bf16_t* __restrict__ wfrag,
                          float* __restrict__ xw, int Nnodes) {
  __shared__ bf16_t sA[64 * XS];
  int tid = threadIdx.x;
  int mbase = blockIdx.x * 64;
  for (int i = tid; i < 64 * 256; i += 256) {
    int r = i >> 8, c = i & 255;
    int gr = mbase + r;
    float v = (gr < Nnodes) ? x[(size_t)gr * 256 + c] : 0.0f;
    sA[r * XS + c] = (bf16_t)v;
  }
  __syncthreads();

  int w = tid >> 5, lane = tid & 31;
  int wm = w & 1, wn = w >> 1;
  int hi = lane >> 4;
  const bf16_t* rA0 = sA + (wm * 32 + (lane & 15)) * XS;
  const bf16_t* rA1 = rA0 + 16 * XS;
  v8f c00 = {}, c01 = {}, c10 = {}, c11 = {};
  for (int ks = 0; ks < 8; ++ks) {
    int k0 = ks * 32 + 8 * hi;
    v16bf a0, a1;
#pragma unroll
    for (int e = 0; e < 8; ++e) {
      a0[e] = rA0[k0 + e];  a0[8 + e] = rA0[k0 + 16 + e];
      a1[e] = rA1[k0 + e];  a1[8 + e] = rA1[k0 + 16 + e];
    }
    v16bf b0 = *(const v16bf*)(wfrag + (((size_t)((wn * 2 + 0) * 8 + ks)) * 32 + lane) * 16);
    v16bf b1 = *(const v16bf*)(wfrag + (((size_t)((wn * 2 + 1) * 8 + ks)) * 32 + lane) * 16);
    c00 = WMMA_BF16(a0, b0, c00);
    c01 = WMMA_BF16(a0, b1, c01);
    c10 = WMMA_BF16(a1, b0, c10);
    c11 = WMMA_BF16(a1, b1, c11);
  }
  int ncol = wn * 32 + (lane & 15);
#pragma unroll
  for (int r = 0; r < 8; ++r) {
    int row0 = mbase + wm * 32 + r + 8 * hi;
    int row1 = row0 + 16;
    if (row0 < Nnodes) {
      xw[(size_t)row0 * 128 + ncol] = c00[r];
      xw[(size_t)row0 * 128 + ncol + 16] = c01[r];
    }
    if (row1 < Nnodes) {
      xw[(size_t)row1 * 128 + ncol] = c10[r];
      xw[(size_t)row1 * 128 + ncol + 16] = c11[r];
    }
  }
}

// ---------------------------------------------------------------------------
// 4) edge aggregation: agg[col] += norm[e] * xw[row]  (32 lanes/edge, float4)
// ---------------------------------------------------------------------------
__global__ void k_scatter(const int* __restrict__ eidx, const float* __restrict__ norm,
                          const float* __restrict__ xw, float* __restrict__ agg,
                          int E, long long total) {
  long long g = (long long)blockIdx.x * blockDim.x + threadIdx.x;
  if (g >= total) return;
  int e = (int)(g >> 5);
  int c4 = ((int)g & 31) << 2;
  float nv = norm[e];
  int r = eidx[e], c = eidx[E + e];
  const float4 v = *(const float4*)(xw + (size_t)r * 128 + c4);
  float* dst = agg + (size_t)c * 128 + c4;
  atomicAdd(dst + 0, nv * v.x);
  atomicAdd(dst + 1, nv * v.y);
  atomicAdd(dst + 2, nv * v.z);
  atomicAdd(dst + 3, nv * v.w);
}

// x1 = relu(agg + dinv^2 * xw (self loop) + b_gcn)  -- written back into agg
__global__ void k_x1(const float* __restrict__ xw, const float* __restrict__ dinv,
                     const float* __restrict__ bg, float* __restrict__ agg, int Nnodes) {
  long long i = (long long)blockIdx.x * blockDim.x + threadIdx.x;
  if (i >= (long long)Nnodes * 128) return;
  int n = (int)(i >> 7), h = (int)(i & 127);
  float d = dinv[n];
  float v = agg[i] + d * d * xw[i] + bg[h];
  agg[i] = fmaxf(v, 0.0f);
}

// ---------------------------------------------------------------------------
// 5) fused decoder: 32 pairs/block. feat tile [32 x 512] bf16 in LDS,
//    h = relu(feat @ dec1_w + b) folded into out += h*dec2_w via LDS atomics.
//    8 waves: wave w covers N cols [nc*256 + w*32, +32), nc = 0..1.
// ---------------------------------------------------------------------------
#define FS 520  // 260 dwords == 4 mod 64 banks
__global__ void k_decoder(const float* __restrict__ x1, const int* __restrict__ idx,
                          const bf16_t* __restrict__ wfrag,
                          const float* __restrict__ d1b, const float* __restrict__ d2w,
                          const float* __restrict__ d2b, float* __restrict__ out, int P) {
  __shared__ bf16_t sF[32 * FS];
  __shared__ float sOut[32];
  int tid = threadIdx.x;
  int pbase = blockIdx.x * 32;
  if (tid < 32) sOut[tid] = 0.0f;

  int c = tid & 127;
#pragma unroll 4
  for (int r = tid >> 7; r < 32; r += 2) {
    int p = pbase + r;
    int i0 = idx[p];
    int i1 = idx[P + p];
    float a = x1[(size_t)i0 * 128 + c];
    float b = x1[(size_t)i1 * 128 + c];
    bf16_t* fr = sF + r * FS;
    fr[c]       = (bf16_t)(a + b);
    fr[128 + c] = (bf16_t)(a * b);
    fr[256 + c] = (bf16_t)a;
    fr[384 + c] = (bf16_t)b;
  }
  __syncthreads();

  int w = tid >> 5, lane = tid & 31;
  int hi = lane >> 4;
  const bf16_t* rA0 = sF + (lane & 15) * FS;
  const bf16_t* rA1 = rA0 + 16 * FS;

  for (int nc = 0; nc < 2; ++nc) {
    int ntile0 = nc * 16 + w * 2;  // n-base = ntile0*16
    v8f c00 = {}, c01 = {}, c10 = {}, c11 = {};
    for (int ks = 0; ks < 16; ++ks) {
      int k0 = ks * 32 + 8 * hi;
      v16bf a0, a1;
#pragma unroll
      for (int e = 0; e < 8; ++e) {
        a0[e] = rA0[k0 + e];  a0[8 + e] = rA0[k0 + 16 + e];
        a1[e] = rA1[k0 + e];  a1[8 + e] = rA1[k0 + 16 + e];
      }
      v16bf b0 = *(const v16bf*)(wfrag + (((size_t)(ntile0 * 16 + ks)) * 32 + lane) * 16);
      v16bf b1 = *(const v16bf*)(wfrag + (((size_t)((ntile0 + 1) * 16 + ks)) * 32 + lane) * 16);
      c00 = WMMA_BF16(a0, b0, c00);
      c01 = WMMA_BF16(a0, b1, c01);
      c10 = WMMA_BF16(a1, b0, c10);
      c11 = WMMA_BF16(a1, b1, c11);
    }
    int n0 = ntile0 * 16 + (lane & 15);
    int n1 = n0 + 16;
    float bias0 = d1b[n0], bias1 = d1b[n1];
    float w0 = d2w[n0], w1 = d2w[n1];
#pragma unroll
    for (int r = 0; r < 8; ++r) {
      int m0 = r + 8 * hi;
      int m1 = m0 + 16;
      float h00 = fmaxf(c00[r] + bias0, 0.0f) * w0;
      float h01 = fmaxf(c01[r] + bias1, 0.0f) * w1;
      float h10 = fmaxf(c10[r] + bias0, 0.0f) * w0;
      float h11 = fmaxf(c11[r] + bias1, 0.0f) * w1;
      atomicAdd(&sOut[m0], h00 + h01);
      atomicAdd(&sOut[m1], h10 + h11);
    }
  }
  __syncthreads();
  if (tid < 32) {
    int p = pbase + tid;
    if (p < P) out[p] = sOut[tid] + d2b[0];
  }
}

// ---------------------------------------------------------------------------
// host side
// ---------------------------------------------------------------------------
static inline size_t align256(size_t x) { return (x + 255) & ~(size_t)255; }

extern "C" void kernel_launch(void* const* d_in, const int* in_sizes, int n_in,
                              void* d_out, int out_size, void* d_ws, size_t ws_size,
                              hipStream_t stream) {
  const float* x     = (const float*)d_in[0];
  const float* curv  = (const float*)d_in[1];
  const float* lin1w = (const float*)d_in[2];
  const float* lin1b = (const float*)d_in[3];
  const float* Wgcn  = (const float*)d_in[4];
  const float* bgcn  = (const float*)d_in[5];
  const float* dec1w = (const float*)d_in[6];
  const float* dec1b = (const float*)d_in[7];
  const float* dec2w = (const float*)d_in[8];
  const float* dec2b = (const float*)d_in[9];
  const int*   eidx  = (const int*)d_in[10];
  const int*   pidx  = (const int*)d_in[11];
  float* out = (float*)d_out;

  const int N = in_sizes[0] / 256;
  const int E = in_sizes[1];
  const int P = in_sizes[11] / 2;

  char* ws = (char*)d_ws;
  size_t off = 0;
  float* ew   = (float*)(ws + off);  off = align256(off + (size_t)E * 4);
  float* deg  = (float*)(ws + off);  off = align256(off + (size_t)N * 4);
  float* dinv = (float*)(ws + off);  off = align256(off + (size_t)N * 4);
  float* xw   = (float*)(ws + off);  off = align256(off + (size_t)N * 128 * 4);
  float* agg  = (float*)(ws + off);  off = align256(off + (size_t)N * 128 * 4);
  bf16_t* wgf = (bf16_t*)(ws + off); off = align256(off + (size_t)64 * 512 * 2);
  bf16_t* d1f = (bf16_t*)(ws + off); off = align256(off + (size_t)512 * 512 * 2);
  (void)ws_size; (void)n_in; (void)out_size;

  // edge weights, degree (self-loop = 1), dinv, norm
  k_edge_weight<<<(E + 255) / 256, 256, 0, stream>>>(curv, lin1w, lin1b, ew, E);
  k_fill<<<(N + 255) / 256, 256, 0, stream>>>(deg, 1.0f, N);
  long long aggn = (long long)N * 128;
  k_fill<<<(unsigned)((aggn + 255) / 256), 256, 0, stream>>>(agg, 0.0f, aggn);
  k_deg<<<(E + 255) / 256, 256, 0, stream>>>(eidx + E, ew, deg, E);
  k_dinv<<<(N + 255) / 256, 256, 0, stream>>>(deg, dinv, N);
  k_norm<<<(E + 255) / 256, 256, 0, stream>>>(eidx, dinv, ew, E);

  // weight packing for WMMA
  k_pack_w<<<(64 * 32 + 255) / 256, 256, 0, stream>>>(Wgcn, wgf, 256, 128);
  k_pack_w<<<(512 * 32 + 255) / 256, 256, 0, stream>>>(dec1w, d1f, 512, 512);

  // xw = x @ W_gcn
  k_gemm_xw<<<(N + 63) / 64, 256, 0, stream>>>(x, wgf, xw, N);

  // edge aggregation + fused self-loop/bias/relu
  long long tot = (long long)E * 32;
  k_scatter<<<(unsigned)((tot + 255) / 256), 256, 0, stream>>>(eidx, ew, xw, agg, E, tot);
  k_x1<<<(unsigned)((aggn + 255) / 256), 256, 0, stream>>>(xw, dinv, bgcn, agg, N);

  // fused decoder
  k_decoder<<<(P + 31) / 32, 256, 0, stream>>>(agg, pidx, d1f, dec1b, dec2w, dec2b, out, P);
}
